// RGCN_16604343566600
// MI455X (gfx1250) — matrix operand order repaired and checked by reading, not compiled, not run on hardware
//
#include <hip/hip_runtime.h>
#include <math.h>
#include <stdint.h>

// ---------------------------------------------------------------------------
// Problem constants (match reference)
// ---------------------------------------------------------------------------
constexpr int NU  = 50000;
constexpr int NI  = 20000;
constexpr int NE  = 400000;
constexpr int FIN = 128;
constexpr int HID = 256;
constexpr int NH  = 4;          // heads

// ---------------------------------------------------------------------------
// WMMA types (CDNA5 wave32)
// ---------------------------------------------------------------------------
typedef __bf16 bf16;
typedef bf16  v4bf  __attribute__((ext_vector_type(4)));
typedef bf16  v8bf  __attribute__((ext_vector_type(8)));
typedef bf16  v16bf __attribute__((ext_vector_type(16)));
typedef float v8f   __attribute__((ext_vector_type(8)));

// ---------------------------------------------------------------------------
// GEMM: C[M,N] = A[M,K] * Bt[N,K]^T   (A,Bt pre-converted bf16 in global,
// fp32 WMMA accumulation).  Block = 256 threads = 8 waves.
// Tile BM=128, BN=64, BK=32; wave w owns rows [w*16, w*16+16) x 4 N-subtiles.
//
// LDS staging is done by the CDNA5 async DMA path:
//   global_load_async_to_lds_b128  (ASYNCcnt) + s_wait_asynccnt, double
// buffered: tile t+1 streams into buf[(t+1)&1] while tile t is consumed.
//
// LDS layouts are fragment-ordered so each WMMA operand is two ds_load_b128:
//   As [row][k]   row-stride 40 bf16 (80 B)  A-frag k runs {h*8..}, {16+h*8..}
//   BsT[col][k]   col-stride 40 bf16 (80 B)  B-frag k runs {h*8..}, {16+h*8..}
// 80 B = 20 dwords -> 16 lanes hit 16 distinct banks; 80 % 16 == 0 keeps all
// b128 transactions aligned.
// ---------------------------------------------------------------------------
#define BM 128
#define BN 64
#define BK 32
#define SA 40   // padded row stride (bf16 elems) for As
#define SB 40   // padded col stride (bf16 elems) for BsT

__global__ __launch_bounds__(256)
void gemm_wmma_bf16(const bf16* __restrict__ A, const bf16* __restrict__ Bt,
                    float* __restrict__ C, int M, int K, int N)
{
    __shared__ __align__(16) bf16 shA[2][BM * SA];
    __shared__ __align__(16) bf16 shB[2][BN * SB];

    const int tid  = threadIdx.x;
    const int wave = tid >> 5;
    const int lane = tid & 31;
    const int hlf  = lane >> 4;     // 16-lane half
    const int l16  = lane & 15;

    const int mBase = blockIdx.y * BM;
    const int nBase = blockIdx.x * BN;

    const unsigned ldsA0 = (unsigned)(uintptr_t)&shA[0][0];
    const unsigned ldsA1 = (unsigned)(uintptr_t)&shA[1][0];
    const unsigned ldsB0 = (unsigned)(uintptr_t)&shB[0][0];
    const unsigned ldsB1 = (unsigned)(uintptr_t)&shB[1][0];

    // Per-thread async-DMA issue for one K-step tile into buffer `buf`.
    auto issueTile = [&](int k0, int buf) {
        const unsigned lA = buf ? ldsA1 : ldsA0;
        const unsigned lB = buf ? ldsB1 : ldsB0;
        // A tile: 128 rows x 32 k = 512 x 16B segments, 2 per thread.
        #pragma unroll
        for (int i = 0; i < 2; ++i) {
            int si  = i * 256 + tid;
            int r   = si >> 2;            // row in tile
            int seg = si & 3;             // 8-elem (16 B) segment in row
            int row = mBase + r; if (row >= M) row = M - 1;   // clamp: row discarded later
            unsigned long long ga = (unsigned long long)(uintptr_t)
                                    (A + (long long)row * K + k0 + seg * 8);
            unsigned la = lA + (unsigned)(r * (SA * 2) + seg * 16);
            asm volatile("global_load_async_to_lds_b128 %0, %1, off"
                         :: "v"(la), "v"(ga) : "memory");
        }
        // B tile: 64 cols x 32 k = 256 x 16B segments, 1 per thread.
        {
            int n   = tid >> 2;
            int seg = tid & 3;
            unsigned long long gb = (unsigned long long)(uintptr_t)
                                    (Bt + (long long)(nBase + n) * K + k0 + seg * 8);
            unsigned lb = lB + (unsigned)(n * (SB * 2) + seg * 16);
            asm volatile("global_load_async_to_lds_b128 %0, %1, off"
                         :: "v"(lb), "v"(gb) : "memory");
        }
    };

    v8f acc[4];
    #pragma unroll
    for (int s = 0; s < 4; ++s)
        #pragma unroll
        for (int r = 0; r < 8; ++r) acc[s][r] = 0.0f;

    const int nT = K / BK;
    issueTile(0, 0);

    for (int t = 0; t < nT; ++t) {
        // make this wave's DMA visible, then rendezvous with the other waves
        asm volatile("s_wait_asynccnt 0x0" ::: "memory");
        __syncthreads();

        if (t + 1 < nT) issueTile((t + 1) * BK, (t + 1) & 1);

        const bf16* As  = (t & 1) ? shA[1] : shA[0];
        const bf16* BsT = (t & 1) ? shB[1] : shB[0];

        // A fragment: row = wave*16 + l16; two contiguous 8-elem runs.
        const bf16* aRow = &As[(wave * 16 + l16) * SA];
        v8bf a0 = *(const v8bf*)&aRow[hlf * 8];
        v8bf a1 = *(const v8bf*)&aRow[16 + hlf * 8];
        v16bf aF = __builtin_shufflevector(a0, a1,
                     0,1,2,3,4,5,6,7,8,9,10,11,12,13,14,15);

        #pragma unroll
        for (int s = 0; s < 4; ++s) {
            const bf16* bCol = &BsT[(s * 16 + l16) * SB + hlf * 8];
            v8bf b0 = *(const v8bf*)&bCol[0];
            v8bf b1 = *(const v8bf*)&bCol[16];
            v16bf bF = __builtin_shufflevector(b0, b1,
                         0,1,2,3,4,5,6,7,8,9,10,11,12,13,14,15);

            acc[s] = __builtin_amdgcn_wmma_f32_16x16x32_bf16(
                         false, aF, false, bF, (short)0, acc[s], false, false);
        }
    }

    // C/D layout: lanes 0-15 -> M = r, lanes 16-31 -> M = r + 8 ; N = l16
    #pragma unroll
    for (int s = 0; s < 4; ++s) {
        int col = nBase + s * 16 + l16;
        #pragma unroll
        for (int r = 0; r < 8; ++r) {
            int row = mBase + wave * 16 + hlf * 8 + r;
            if (row < M && col < N)
                C[(long long)row * N + col] = acc[s][r];
        }
    }
}

// ---------------------------------------------------------------------------
// fp32 -> bf16 conversion (n multiple of 4) and fused transpose-convert
// ---------------------------------------------------------------------------
__global__ void cvt_bf16(const float* __restrict__ x, bf16* __restrict__ y,
                         long long nQuads)
{
    long long t = blockIdx.x * (long long)blockDim.x + threadIdx.x;
    if (t >= nQuads) return;
    const float4 f = *(const float4*)&x[t * 4];
    v4bf vb; vb[0] = (bf16)f.x; vb[1] = (bf16)f.y;
             vb[2] = (bf16)f.z; vb[3] = (bf16)f.w;
    *(v4bf*)&y[t * 4] = vb;
}

__global__ void cvt_transpose_bf16(const float* __restrict__ W,  // [K,N]
                                   bf16* __restrict__ Wt,        // [N,K]
                                   int K, int N)
{
    int t = blockIdx.x * blockDim.x + threadIdx.x;
    if (t >= K * N) return;
    int k = t / N, n = t - k * N;
    Wt[(long long)n * K + k] = (bf16)W[t];
}

// ---------------------------------------------------------------------------
// el/er: out[n,h] = sum_d h[n, h*dh + d] * Avec[h*dh + d]
// ---------------------------------------------------------------------------
__global__ void attn_scores(const float* __restrict__ h,
                            const float* __restrict__ Avec,
                            float* __restrict__ out, int N, int dh)
{
    int t = blockIdx.x * blockDim.x + threadIdx.x;     // n*NH + head
    if (t >= N * NH) return;
    int n = t >> 2, hh = t & 3;
    const float* row = h + (long long)n * NH * dh + hh * dh;
    const float* a   = Avec + hh * dh;
    float s = 0.0f;
    for (int d = 0; d < dh; ++d) s += row[d] * a[d];
    out[t] = s;
}

__global__ void fill_val(float* p, float v, long long n)
{
    long long t = blockIdx.x * (long long)blockDim.x + threadIdx.x;
    if (t < n) p[t] = v;
}

__device__ inline void atomicMaxF(float* addr, float val)
{
    unsigned int* ia = (unsigned int*)addr;
    unsigned int old = *ia;
    while (__uint_as_float(old) < val) {
        unsigned int assumed = old;
        old = atomicCAS(ia, assumed, __float_as_uint(val));
        if (old == assumed) break;
    }
}

// e = leaky_relu(el[src]+er[dst], 0.2); segment max over dst
__global__ void edge_logits(const float* __restrict__ el,
                            const float* __restrict__ er,
                            const int* __restrict__ src,
                            const int* __restrict__ dst,
                            float* __restrict__ eSto, float* __restrict__ mBuf,
                            int nE)
{
    int t = blockIdx.x * blockDim.x + threadIdx.x;
    if (t >= nE * NH) return;
    int e = t >> 2, hh = t & 3;
    float v = el[src[e] * NH + hh] + er[dst[e] * NH + hh];
    v = (v > 0.0f) ? v : 0.2f * v;
    eSto[t] = v;
    atomicMaxF(&mBuf[dst[e] * NH + hh], v);
}

// ee = exp(e - max); segment sum over dst
__global__ void edge_exp(float* __restrict__ eSto,
                         const float* __restrict__ mBuf,
                         float* __restrict__ dsum,
                         const int* __restrict__ dst, int nE)
{
    int t = blockIdx.x * blockDim.x + threadIdx.x;
    if (t >= nE * NH) return;
    int e = t >> 2, hh = t & 3;
    float v = expf(eSto[t] - mBuf[dst[e] * NH + hh]);
    eSto[t] = v;
    atomicAdd(&dsum[dst[e] * NH + hh], v);
}

// out[dst] += hs[src] * alpha      (one block per edge, one thread per feature)
__global__ void edge_scatter(const float* __restrict__ hs,
                             const float* __restrict__ eSto,
                             const float* __restrict__ dsum,
                             const int* __restrict__ src,
                             const int* __restrict__ dst,
                             float* __restrict__ out, int dh)
{
    int e  = blockIdx.x;
    int f  = threadIdx.x;            // 0 .. NH*dh-1
    int hh = f / dh;
    int s  = src[e], d = dst[e];
    float alpha = eSto[e * NH + hh] / dsum[d * NH + hh];
    atomicAdd(&out[(long long)d * (NH * dh) + f],
              hs[(long long)s * (NH * dh) + f] * alpha);
}

// ---------------------------------------------------------------------------
// BatchNorm (biased var) + activation: act 0 = leaky_relu(0.01), 1 = tanh
// ---------------------------------------------------------------------------
__global__ void bn_stats(const float* __restrict__ x, float* __restrict__ sums,
                         int N, int C)
{
    long long total = (long long)N * C;
    long long stride = (long long)gridDim.x * blockDim.x;
    for (long long t = blockIdx.x * (long long)blockDim.x + threadIdx.x;
         t < total; t += stride) {
        int c = (int)(t % C);
        float v = x[t];
        atomicAdd(&sums[c], v);
        atomicAdd(&sums[C + c], v * v);
    }
}

__global__ void bn_apply(const float* __restrict__ x, float* __restrict__ y,
                         const float* __restrict__ sums,
                         const float* __restrict__ gamma,
                         const float* __restrict__ beta,
                         int N, int C, int act)
{
    long long t = blockIdx.x * (long long)blockDim.x + threadIdx.x;
    if (t >= (long long)N * C) return;
    int c = (int)(t % C);
    float mean = sums[c] / (float)N;
    float var  = sums[C + c] / (float)N - mean * mean;
    float v = gamma[c] * (x[t] - mean) * rsqrtf(var + 1e-5f) + beta[c];
    y[t] = (act == 1) ? tanhf(v) : ((v > 0.0f) ? v : 0.01f * v);
}

// ---------------------------------------------------------------------------
// Host side: one bipartite GATConv (activations pre-converted to bf16)
// ---------------------------------------------------------------------------
static void run_gat(hipStream_t stream,
                    const bf16* hsrcB, const bf16* hdstB, int fin,
                    const float* Wmat, bf16* wtB,        // fp32 W, bf16 Wt scratch
                    const float* Amat, int dh,
                    const int* src, const int* dst, int nSrc, int nDst,
                    float* out,                          // [nDst, NH*dh]
                    float* psS, float* psD, float* el, float* er,
                    float* ee, float* mbuf, float* dsum)
{
    const int Nf = NH * dh;

    cvt_transpose_bf16<<<(fin * Nf + 255) / 256, 256, 0, stream>>>(Wmat, wtB, fin, Nf);

    dim3 gs((Nf + BN - 1) / BN, (nSrc + BM - 1) / BM);
    gemm_wmma_bf16<<<gs, 256, 0, stream>>>(hsrcB, wtB, psS, nSrc, fin, Nf);
    dim3 gd((Nf + BN - 1) / BN, (nDst + BM - 1) / BM);
    gemm_wmma_bf16<<<gd, 256, 0, stream>>>(hdstB, wtB, psD, nDst, fin, Nf);

    attn_scores<<<(nSrc * NH + 255) / 256, 256, 0, stream>>>(psS, Amat, el, nSrc, dh);
    attn_scores<<<(nDst * NH + 255) / 256, 256, 0, stream>>>(psD, Amat + Nf, er, nDst, dh);

    fill_val<<<(nDst * NH + 255) / 256, 256, 0, stream>>>(mbuf, -1e30f, (long long)nDst * NH);
    hipMemsetAsync(dsum, 0, (size_t)nDst * NH * sizeof(float), stream);
    hipMemsetAsync(out, 0, (size_t)nDst * Nf * sizeof(float), stream);

    edge_logits<<<(NE * NH + 255) / 256, 256, 0, stream>>>(el, er, src, dst, ee, mbuf, NE);
    edge_exp<<<(NE * NH + 255) / 256, 256, 0, stream>>>(ee, mbuf, dsum, dst, NE);
    edge_scatter<<<NE, Nf, 0, stream>>>(psS, ee, dsum, src, dst, out, dh);
}

// ---------------------------------------------------------------------------
// kernel_launch
// ---------------------------------------------------------------------------
extern "C" void kernel_launch(void* const* d_in, const int* in_sizes, int n_in,
                              void* d_out, int out_size, void* d_ws, size_t ws_size,
                              hipStream_t stream)
{
    const float* x_user = (const float*)d_in[0];
    const float* x_item = (const float*)d_in[1];
    const int*   eu     = (const int*)d_in[2];
    const int*   ei     = (const int*)d_in[3];

    const float* W[4][2];
    const float* Aat[4][2];
    for (int L = 0; L < 4; ++L) {
        W[L][0]   = (const float*)d_in[4 + 4 * L + 0];   // W{L}_ui
        Aat[L][0] = (const float*)d_in[4 + 4 * L + 1];   // A{L}_ui
        W[L][1]   = (const float*)d_in[4 + 4 * L + 2];   // W{L}_iu
        Aat[L][1] = (const float*)d_in[4 + 4 * L + 3];   // A{L}_iu
    }
    const float* bn_u = (const float*)d_in[20];          // [3,2,HID]
    const float* bn_i = (const float*)d_in[21];

    // ---- carve workspace (byte allocator, 256 B aligned blocks) ----
    char* base = (char*)d_ws;
    size_t off = 0;
    auto allocB = [&](size_t bytes) {
        void* p = base + off; off += (bytes + 255) & ~(size_t)255; return p;
    };

    float* huB    = (float*)allocB((size_t)NU * HID * 4);   // activated user feats
    float* hiB    = (float*)allocB((size_t)NI * HID * 4);   // activated item feats
    float* nu_raw = (float*)allocB((size_t)NU * HID * 4);
    float* ni_raw = (float*)allocB((size_t)NI * HID * 4);
    float* psS    = (float*)allocB((size_t)NU * HID * 4);   // src projection (max)
    float* psD    = (float*)allocB((size_t)NU * HID * 4);   // dst projection (max)
    float* el     = (float*)allocB((size_t)NU * NH * 4);
    float* er     = (float*)allocB((size_t)NU * NH * 4);
    float* mbuf   = (float*)allocB((size_t)NU * NH * 4);
    float* dsum   = (float*)allocB((size_t)NU * NH * 4);
    float* ee     = (float*)allocB((size_t)NE * NH * 4);
    float* bns    = (float*)allocB(2 * HID * 4);
    bf16* huBf    = (bf16*)allocB((size_t)NU * HID * 2);    // bf16 activations
    bf16* hiBf    = (bf16*)allocB((size_t)NI * HID * 2);
    bf16* wtBf    = (bf16*)allocB((size_t)HID * HID * 2);   // bf16 W^T scratch
    (void)ws_size; (void)in_sizes; (void)n_in; (void)out_size;

    const float* hu_cur = x_user;
    const float* hi_cur = x_item;
    int fin = FIN;

    for (int L = 0; L < 4; ++L) {
        const int dh = (L == 3) ? 32 : 64;

        // activations -> bf16 (once per layer, feeds 4 GEMMs)
        cvt_bf16<<<((long long)NU * fin / 4 + 255) / 256, 256, 0, stream>>>(
            hu_cur, huBf, (long long)NU * fin / 4);
        cvt_bf16<<<((long long)NI * fin / 4 + 255) / 256, 256, 0, stream>>>(
            hi_cur, hiBf, (long long)NI * fin / 4);

        float* outI = (L == 3) ? (float*)d_out + (size_t)NU * 128 : ni_raw;
        float* outU = (L == 3) ? (float*)d_out                    : nu_raw;

        // user -> item  (src=user, dst=item)
        run_gat(stream, huBf, hiBf, fin, W[L][0], wtBf, Aat[L][0], dh,
                eu, ei, NU, NI, outI, psS, psD, el, er, ee, mbuf, dsum);
        // item -> user  (src=item, dst=user)
        run_gat(stream, hiBf, huBf, fin, W[L][1], wtBf, Aat[L][1], dh,
                ei, eu, NI, NU, outU, psS, psD, el, er, ee, mbuf, dsum);

        if (L == 3) break;

        int act = (L == 2) ? 1 : 0;
        // BN user
        hipMemsetAsync(bns, 0, 2 * HID * sizeof(float), stream);
        bn_stats<<<1024, 256, 0, stream>>>(nu_raw, bns, NU, HID);
        bn_apply<<<((long long)NU * HID + 255) / 256, 256, 0, stream>>>(
            nu_raw, huB, bns, bn_u + (size_t)(L * 2) * HID,
            bn_u + (size_t)(L * 2 + 1) * HID, NU, HID, act);
        // BN item
        hipMemsetAsync(bns, 0, 2 * HID * sizeof(float), stream);
        bn_stats<<<1024, 256, 0, stream>>>(ni_raw, bns, NI, HID);
        bn_apply<<<((long long)NI * HID + 255) / 256, 256, 0, stream>>>(
            ni_raw, hiB, bns, bn_i + (size_t)(L * 2) * HID,
            bn_i + (size_t)(L * 2 + 1) * HID, NI, HID, act);

        hu_cur = huB;
        hi_cur = hiB;
        fin = HID;
    }
}
